// OPTSelfAttention_59124519796946
// MI455X (gfx1250) — matrix operand order repaired and checked by reading, not compiled, not run on hardware
//
#include <hip/hip_runtime.h>

// ---------------------------------------------------------------------------
// OPT self-attention for MI455X (gfx1250): bf16 WMMA GEMM + flash attention.
// B=2, S=2048, D=2048, H=32, HD=64.
// ---------------------------------------------------------------------------

typedef __attribute__((ext_vector_type(16))) __bf16        v16bf;
typedef __attribute__((ext_vector_type(8)))  float         v8f;
typedef __attribute__((ext_vector_type(8)))  unsigned int  v8u;
typedef __attribute__((ext_vector_type(4)))  float         v4f;
typedef __attribute__((ext_vector_type(4)))  unsigned int  v4u;
typedef __attribute__((ext_vector_type(2)))  unsigned int  v2u;

static constexpr int CB  = 2;     // batch
static constexpr int CS  = 2048;  // seq
static constexpr int CD  = 2048;  // model dim
static constexpr int CH  = 32;    // heads
static constexpr int CHD = 64;    // head dim
static constexpr int C3D = 3 * CD;

__device__ __forceinline__ unsigned short f2bf_u(float f) {
    unsigned u = __builtin_bit_cast(unsigned, f);
    u += 0x7FFFu + ((u >> 16) & 1u);          // round-to-nearest-even
    return (unsigned short)(u >> 16);
}
__device__ __forceinline__ unsigned pack2bf(float lo, float hi) {
    return (unsigned)f2bf_u(lo) | ((unsigned)f2bf_u(hi) << 16);
}

// A-fragment K offset for 16-bit 16x32 A matrix (ISA 7.12.2):
// lanes 0-15 hold K {0..7,16..23}, lanes 16-31 hold K {8..15,24..31}
__device__ __forceinline__ int akoff(int i, int hgrp) {
    int k = (i < 4) ? (2 * i) : (16 + 2 * (i - 4));
    return k + hgrp * 8;
}

// ---------------------------------------------------------------------------
// Kernel 1: fused QKV projection  C = hs @ W + b, interleave split, bf16 cast.
// Grid: (4096/64, 6144/64), block 128 threads (4 waves), 64x64 tile / block.
// ---------------------------------------------------------------------------
__global__ __launch_bounds__(128)
void qkv_gemm_kernel(const float* __restrict__ hs,
                     const float* __restrict__ W,
                     const float* __restrict__ bias,
                     unsigned short* __restrict__ Qb,
                     unsigned short* __restrict__ Kb,
                     unsigned short* __restrict__ Vb) {
    __shared__ __align__(16) unsigned short As[64 * 32];  // [row][k]
    __shared__ __align__(16) unsigned short Bs[64 * 32];  // [col][k]

    const int tid  = threadIdx.x;
    const int wave = tid >> 5;
    const int lane = tid & 31;
    const int hgrp = lane >> 4;
    const int l16  = lane & 15;
    const int m0 = blockIdx.x * 64;   // rows of [B*S, D]
    const int n0 = blockIdx.y * 64;   // cols of [D, 3D]

    v8f acc[4] = {};

    for (int k0 = 0; k0 < CD; k0 += 32) {
        // prefetch next K-step tiles into cache (global_prefetch_b8)
        if (k0 + 32 < CD) {
            __builtin_prefetch(&hs[(size_t)(m0 + (tid >> 1)) * CD + k0 + 32 + (tid & 1) * 16], 0, 1);
            __builtin_prefetch(&W[(size_t)(k0 + 32 + (tid >> 2)) * C3D + n0 + (tid & 3) * 16], 0, 1);
        }

        // stage A tile (64x32) bf16 row-major: b128 loads, packed dword stores
#pragma unroll
        for (int it = 0; it < 4; ++it) {
            int i  = tid + it * 128;          // 0..511
            int r  = i >> 3;                  // 0..63
            int c4 = (i & 7) * 4;             // 0..28 step 4
            v4f f = *(const v4f*)&hs[(size_t)(m0 + r) * CD + k0 + c4];
            *(unsigned*)&As[r * 32 + c4]     = pack2bf(f[0], f[1]);
            *(unsigned*)&As[r * 32 + c4 + 2] = pack2bf(f[2], f[3]);
        }
        // stage B tile (32x64) -> Bs[col][k]: register pack-transpose of k-pairs
#pragma unroll
        for (int it = 0; it < 2; ++it) {
            int i  = tid + it * 128;          // 0..255
            int kp = i >> 4;                  // k-pair, kk = 2*kp
            int cg = i & 15;                  // col group of 4
            v4f f0 = *(const v4f*)&W[(size_t)(k0 + 2 * kp)     * C3D + n0 + cg * 4];
            v4f f1 = *(const v4f*)&W[(size_t)(k0 + 2 * kp + 1) * C3D + n0 + cg * 4];
#pragma unroll
            for (int c = 0; c < 4; ++c)
                *(unsigned*)&Bs[(cg * 4 + c) * 32 + 2 * kp] = pack2bf(f0[c], f1[c]);
        }
        __syncthreads();

        // A fragment: rows [wave*16, wave*16+16)
        v8u au;
#pragma unroll
        for (int i = 0; i < 8; ++i)
            au[i] = *(const unsigned int*)&As[(wave * 16 + l16) * 32 + akoff(i, hgrp)];
        v16bf afrag = __builtin_bit_cast(v16bf, au);

#pragma unroll
        for (int j = 0; j < 4; ++j) {
            v8u bu;
#pragma unroll
            for (int i = 0; i < 8; ++i)
                bu[i] = *(const unsigned int*)&Bs[(j * 16 + l16) * 32 + akoff(i, hgrp)];
            acc[j] = __builtin_amdgcn_wmma_f32_16x16x32_bf16(
                false, afrag, false, __builtin_bit_cast(v16bf, bu),
                (short)0, acc[j], false, false);
        }
        __syncthreads();
    }

    // Epilogue: bias, interleave decode (j = d*3 + ch; ch 0=q,1=v,2=k),
    // fold 1/sqrt(64) into Q, scatter bf16 into [B,H,S,HD] tensors.
    const float qscale = 0.125f;
#pragma unroll
    for (int j = 0; j < 4; ++j) {
#pragma unroll
        for (int r = 0; r < 8; ++r) {
            int Ml = wave * 16 + r + hgrp * 8;        // C-layout row
            int R  = m0 + Ml;                         // global row in [0,4096)
            int Cg = n0 + j * 16 + l16;               // global col in [0,6144)
            float val = acc[j][r] + bias[Cg];
            int d  = Cg / 3, ch = Cg - 3 * d;
            int hh = d >> 6, hd = d & 63;
            int s  = R & (CS - 1), batch = R >> 11;
            size_t idx = (((size_t)(batch * CH + hh)) * CS + s) * CHD + hd;
            if (ch == 0)      Qb[idx] = f2bf_u(val * qscale);
            else if (ch == 1) Vb[idx] = f2bf_u(val);
            else              Kb[idx] = f2bf_u(val);
        }
    }
}

// ---------------------------------------------------------------------------
// Kernel 2: causal flash attention per (b,h). Grid: (S/64, B*H), 4 waves.
// Each wave owns 16 query rows; 32-key blocks; online softmax; fp32 O acc.
// ---------------------------------------------------------------------------
__global__ __launch_bounds__(128)
void flash_attn_kernel(const unsigned short* __restrict__ Qb,
                       const unsigned short* __restrict__ Kb,
                       const unsigned short* __restrict__ Vb,
                       float* __restrict__ out) {
    __shared__ __align__(16) unsigned short Ks[32 * 64];   // [key][hd]
    __shared__ __align__(16) unsigned short Vt[64 * 32];   // [hd][key]
    __shared__ __align__(16) unsigned short Ps[4][16 * 32]; // per-wave P

    const int tid  = threadIdx.x;
    const int wave = tid >> 5;
    const int lane = tid & 31;
    const int hgrp = lane >> 4;
    const int l16  = lane & 15;
    const int bh    = blockIdx.y;
    const int batch = bh >> 5;
    const int hh    = bh & 31;
    const int qbase = blockIdx.x * 64;
    const size_t bhbase = (size_t)bh * CS * CHD;

    // Q fragments, held in registers across the whole key loop.
    const int qrow = qbase + wave * 16 + l16;
    v8u qu0, qu1;
#pragma unroll
    for (int i = 0; i < 8; ++i) {
        int hd = akoff(i, hgrp);
        qu0[i] = *(const unsigned int*)&Qb[bhbase + (size_t)qrow * CHD + hd];
        qu1[i] = *(const unsigned int*)&Qb[bhbase + (size_t)qrow * CHD + hd + 32];
    }
    v16bf qf0 = __builtin_bit_cast(v16bf, qu0);
    v16bf qf1 = __builtin_bit_cast(v16bf, qu1);

    v8f oacc[4] = {};
    float mrow[8], lrow[8];
#pragma unroll
    for (int r = 0; r < 8; ++r) { mrow[r] = -1e30f; lrow[r] = 0.0f; }

    const int qmax = qbase + 63;                 // causal: skip blocks past tile
    for (int kbase = 0; kbase <= qmax; kbase += 32) {
        // K block: contiguous b128 copy (32 keys x 64 hd = 256 x 16B)
        {
            const v4u* src = (const v4u*)(Kb + bhbase + (size_t)kbase * CHD);
            v4u* dst = (v4u*)Ks;
#pragma unroll
            for (int it = 0; it < 2; ++it)
                dst[tid + it * 128] = src[tid + it * 128];
        }
        // V block -> Vt[hd][key]: register pack-transpose of key-pairs
#pragma unroll
        for (int it = 0; it < 2; ++it) {
            int i  = tid + it * 128;          // 0..255
            int kp = i >> 4;                  // key pair, key = 2*kp
            int hg = i & 15;                  // hd group of 4
            const unsigned short* p0 = Vb + bhbase + (size_t)(kbase + 2 * kp) * CHD + hg * 4;
            v2u a0 = *(const v2u*)p0;         // key 2kp,   hd hg*4..+3
            v2u a1 = *(const v2u*)(p0 + CHD); // key 2kp+1, hd hg*4..+3
            *(unsigned*)&Vt[(hg * 4 + 0) * 32 + 2 * kp] = (a0[0] & 0xffffu) | (a1[0] << 16);
            *(unsigned*)&Vt[(hg * 4 + 1) * 32 + 2 * kp] = (a0[0] >> 16) | (a1[0] & 0xffff0000u);
            *(unsigned*)&Vt[(hg * 4 + 2) * 32 + 2 * kp] = (a0[1] & 0xffffu) | (a1[1] << 16);
            *(unsigned*)&Vt[(hg * 4 + 3) * 32 + 2 * kp] = (a0[1] >> 16) | (a1[1] & 0xffff0000u);
        }
        __syncthreads();

        // S = Q @ K^T : two 16x16 key tiles, HD=64 split into 2 chunks of 32
        v8f sacc[2];
#pragma unroll
        for (int j = 0; j < 2; ++j) {
            v8u bu0, bu1;
#pragma unroll
            for (int i = 0; i < 8; ++i) {
                int kk = akoff(i, hgrp);                      // hd pair
                bu0[i] = *(const unsigned int*)&Ks[(j * 16 + l16) * 64 + kk];
                bu1[i] = *(const unsigned int*)&Ks[(j * 16 + l16) * 64 + 32 + kk];
            }
            v8f z = {};
            sacc[j] = __builtin_amdgcn_wmma_f32_16x16x32_bf16(
                false, qf0, false, __builtin_bit_cast(v16bf, bu0), (short)0, z, false, false);
            sacc[j] = __builtin_amdgcn_wmma_f32_16x16x32_bf16(
                false, qf1, false, __builtin_bit_cast(v16bf, bu1), (short)0, sacc[j], false, false);
        }

        // causal mask + online softmax (rows live in 16-lane groups)
        float p0v[8], p1v[8], alpha[8];
#pragma unroll
        for (int r = 0; r < 8; ++r) {
            int qg  = qbase + wave * 16 + r + hgrp * 8;
            int k0g = kbase + l16;
            int k1g = kbase + 16 + l16;
            float s0 = (k0g > qg) ? -1e30f : sacc[0][r];
            float s1 = (k1g > qg) ? -1e30f : sacc[1][r];
            float m = fmaxf(s0, s1);
            m = fmaxf(m, __shfl_xor(m, 1, 32));
            m = fmaxf(m, __shfl_xor(m, 2, 32));
            m = fmaxf(m, __shfl_xor(m, 4, 32));
            m = fmaxf(m, __shfl_xor(m, 8, 32));
            float mnew = fmaxf(mrow[r], m);
            alpha[r] = __expf(mrow[r] - mnew);
            float p0 = __expf(s0 - mnew);
            float p1 = __expf(s1 - mnew);
            float rs = p0 + p1;
            rs += __shfl_xor(rs, 1, 32);
            rs += __shfl_xor(rs, 2, 32);
            rs += __shfl_xor(rs, 4, 32);
            rs += __shfl_xor(rs, 8, 32);
            lrow[r] = lrow[r] * alpha[r] + rs;
            mrow[r] = mnew;
            p0v[r] = p0; p1v[r] = p1;
        }

        // rescale running output
#pragma unroll
        for (int c = 0; c < 4; ++c) {
#pragma unroll
            for (int r = 0; r < 8; ++r) oacc[c][r] *= alpha[r];
        }

        // P: C-layout -> LDS row-major [row][key0..31] -> A-fragment
#pragma unroll
        for (int r = 0; r < 8; ++r) {
            int Ml = r + hgrp * 8;
            Ps[wave][Ml * 32 + l16]      = f2bf_u(p0v[r]);
            Ps[wave][Ml * 32 + 16 + l16] = f2bf_u(p1v[r]);
        }
        v8u pu;
#pragma unroll
        for (int i = 0; i < 8; ++i)
            pu[i] = *(const unsigned int*)&Ps[wave][l16 * 32 + akoff(i, hgrp)];
        v16bf pf = __builtin_bit_cast(v16bf, pu);

        // O += P @ V : 4 hd-chunks of 16 columns
#pragma unroll
        for (int c = 0; c < 4; ++c) {
            v8u bu;
#pragma unroll
            for (int i = 0; i < 8; ++i)
                bu[i] = *(const unsigned int*)&Vt[(c * 16 + l16) * 32 + akoff(i, hgrp)];
            oacc[c] = __builtin_amdgcn_wmma_f32_16x16x32_bf16(
                false, pf, false, __builtin_bit_cast(v16bf, bu), (short)0, oacc[c], false, false);
        }
        __syncthreads();
    }

    // normalize and write fp32 output [B,S,H*HD]
#pragma unroll
    for (int c = 0; c < 4; ++c) {
#pragma unroll
        for (int r = 0; r < 8; ++r) {
            int srow = qbase + wave * 16 + r + hgrp * 8;
            int hd = c * 16 + l16;
            out[((size_t)(batch * CS + srow)) * CD + hh * CHD + hd] = oacc[c][r] / lrow[r];
        }
    }
}

// ---------------------------------------------------------------------------
extern "C" void kernel_launch(void* const* d_in, const int* in_sizes, int n_in,
                              void* d_out, int out_size, void* d_ws, size_t ws_size,
                              hipStream_t stream) {
    const float* hs   = (const float*)d_in[0];
    const float* W    = (const float*)d_in[1];
    const float* bias = (const float*)d_in[2];
    float* out = (float*)d_out;

    const size_t per = (size_t)CB * CH * CS * CHD;   // 8,388,608 elems
    unsigned short* Qb = (unsigned short*)d_ws;
    unsigned short* Kb = Qb + per;
    unsigned short* Vb = Kb + per;

    dim3 g1((CB * CS) / 64, C3D / 64), b1(128);
    qkv_gemm_kernel<<<g1, b1, 0, stream>>>(hs, W, bias, Qb, Kb, Vb);

    dim3 g2(CS / 64, CB * CH), b2(128);
    flash_attn_kernel<<<g2, b2, 0, stream>>>(Qb, Kb, Vb, out);
}
